// SubtractionTransformerBlock_77841987272925
// MI455X (gfx1250) — matrix operand order repaired and checked by reading, not compiled, not run on hardware
//
#include <hip/hip_runtime.h>
#include <hip/hip_bf16.h>
#include <hip/hip_fp16.h>

// Problem constants (match reference)
#define N_NODES 65536
#define N_EDGES 1048576
#define CC      64

typedef _Float16 half8  __attribute__((ext_vector_type(8)));
typedef _Float16 half16 __attribute__((ext_vector_type(16)));
typedef float    vf8    __attribute__((ext_vector_type(8)));

static __device__ __forceinline__ vf8 wmma16(half16 a, half16 b, vf8 c) {
  // D = A(16x32 f16) * B(32x16 f16) + C(16x16 f32)
  return __builtin_amdgcn_wmma_f32_16x16x32_f16(false, a, false, b, (short)0, c,
                                                false, false);
}

// ---- CDNA5 async LDS<->global DMA (ASYNCcnt-tracked), inline asm per ISA 10.x ----
static __device__ __forceinline__ void async_load_b128(unsigned lds_off, const void* gaddr) {
  // LDS[lds_off .. +15] = MEM[gaddr .. +15]
  asm volatile("global_load_async_to_lds_b128 %0, %1, off"
               :: "v"(lds_off), "v"(gaddr) : "memory");
}
static __device__ __forceinline__ void async_store_b128(const void* gaddr, unsigned lds_off) {
  // MEM[gaddr .. +15] = LDS[lds_off .. +15]
  asm volatile("global_store_async_from_lds_b128 %0, %1, off"
               :: "v"(gaddr), "v"(lds_off) : "memory");
}
static __device__ __forceinline__ void async_wait0() {
  asm volatile("s_wait_asynccnt 0" ::: "memory");
}
static __device__ __forceinline__ unsigned lds_offset(const void* p) {
  // generic pointer to LDS: low 32 bits are the workgroup-relative LDS byte offset
  return (unsigned)(size_t)p;
}

// Packed weight fragments: wpk[mat][ks][nt][lane][q], 16 halves per lane (32B load)
static __device__ __forceinline__ half16 wfrag(const _Float16* wpk, int mat, int ks,
                                               int nt, int lane) {
  return *(const half16*)(wpk + ((((mat * 2 + ks) * 4 + nt) * 32 + lane) << 4));
}
// Same layout but served out of LDS (mat2 = 0..2)
static __device__ __forceinline__ half16 wfrag_lds(const _Float16* wl, int mat2, int ks,
                                                   int nt, int lane) {
  return *(const half16*)(wl + ((((mat2 * 2 + ks) * 4 + nt) * 32 + lane) << 4));
}

// A-matrix 16x32 f16 layout (ISA 7.12.2): lane group 0 -> K {0..7,16..23}, group 1 -> {8..15,24..31}
static __device__ __forceinline__ half16 load_a_f16(const _Float16* row, int ks, int lane) {
  const int kg = (lane & 16) ? 8 : 0;
  union { half16 h; half8 p[2]; } u;
  u.p[0] = *(const half8*)(row + ks * 32 + kg);
  u.p[1] = *(const half8*)(row + ks * 32 + 16 + kg);
  return u.h;
}

static __device__ __forceinline__ half16 load_a_lds(const _Float16* base, int ks, int lane) {
  const int m = lane & 15, kg = (lane & 16) ? 8 : 0;
  const _Float16* row = base + m * 72;   // 72-half (144B) padded rows: bank-conflict free
  union { half16 h; half8 p[2]; } u;
  u.p[0] = *(const half8*)(row + ks * 32 + kg);
  u.p[1] = *(const half8*)(row + ks * 32 + 16 + kg);
  return u.h;
}

static __device__ __forceinline__ half16 load_a_f32cvt(const float* row, int ks, int lane) {
  const int kg = (lane & 16) ? 8 : 0;
  half16 r;
#pragma unroll
  for (int q = 0; q < 8; ++q) r[q] = (_Float16)row[ks * 32 + kg + q];
#pragma unroll
  for (int q = 0; q < 8; ++q) r[8 + q] = (_Float16)row[ks * 32 + 16 + kg + q];
  return r;
}

static __device__ __forceinline__ vf8 vzero() {
  vf8 z = {0.f, 0.f, 0.f, 0.f, 0.f, 0.f, 0.f, 0.f};
  return z;
}

// ---------------------------------------------------------------------------
// Kernel 0: repack eight 64x64 f32 weight matrices into f16 B-fragment layout.
// B 32x16 f16 layout: n = lane&15; lanes 0-15 hold K=kbase+{0..7,16..23},
// lanes 16-31 hold K=kbase+{8..15,24..31} (matches A layout).
// mats: 0=W_in 1=W_lin 2=W_src 3=W_dst 4=P2 5=A1 6=A2 7=W_out
// ---------------------------------------------------------------------------
__global__ void prep_weights_kernel(const float* __restrict__ Win, const float* __restrict__ Wlin,
                                    const float* __restrict__ Wsrc, const float* __restrict__ Wdst,
                                    const float* __restrict__ P2,  const float* __restrict__ A1,
                                    const float* __restrict__ A2,  const float* __restrict__ Wout,
                                    _Float16* __restrict__ wpk) {
  int t = blockIdx.x * blockDim.x + threadIdx.x;   // 8 * 4096 threads
  if (t >= 8 * 4096) return;
  int q = t & 15, lane = (t >> 4) & 31, nt = (t >> 9) & 3, ks = (t >> 11) & 1, mat = t >> 12;
  const float* W;
  switch (mat) {
    case 0: W = Win;  break; case 1: W = Wlin; break;
    case 2: W = Wsrc; break; case 3: W = Wdst; break;
    case 4: W = P2;   break; case 5: W = A1;   break;
    case 6: W = A2;   break; default: W = Wout; break;
  }
  int n = nt * 16 + (lane & 15);
  int k = ks * 32 + ((lane & 16) ? 8 : 0) + ((q < 8) ? q : (16 + q - 8));
  wpk[t] = (_Float16)W[k * CC + n];
}

// ---------------------------------------------------------------------------
// Kernel 1: node linears. h = relu(x@W_in + b_in); v = h@W_lin; a_src = h@W_src;
// a_dst = h@W_dst.  One wave per 16-row tile; outputs stored f16 (halves gather BW).
// ---------------------------------------------------------------------------
__global__ void node_in_kernel(const float* __restrict__ x, const _Float16* __restrict__ wpk,
                               const float* __restrict__ b_in,
                               _Float16* __restrict__ vbuf, _Float16* __restrict__ asrc,
                               _Float16* __restrict__ adst) {
  __shared__ __align__(16) _Float16 lds[8][16 * 72];
  const int lane = threadIdx.x & 31, wv = threadIdx.x >> 5;
  const int n0 = (blockIdx.x * 8 + wv) * 16;
  const int m = lane & 15;
  const int col0 = lane & 15, mrow0 = (lane & 16) ? 8 : 0;

  const float* row = x + (size_t)(n0 + m) * CC;
  half16 a0 = load_a_f32cvt(row, 0, lane);
  half16 a1 = load_a_f32cvt(row, 1, lane);

  // h = relu(x @ W_in + b_in) -> LDS (f16, A-reload layout)
#pragma unroll
  for (int nt = 0; nt < 4; ++nt) {
    vf8 acc = vzero();
    acc = wmma16(a0, wfrag(wpk, 0, 0, nt, lane), acc);
    acc = wmma16(a1, wfrag(wpk, 0, 1, nt, lane), acc);
    float b = b_in[nt * 16 + col0];
#pragma unroll
    for (int r = 0; r < 8; ++r) {
      float t = acc[r] + b;
      t = t > 0.f ? t : 0.f;
      lds[wv][(mrow0 + r) * 72 + nt * 16 + col0] = (_Float16)t;
    }
  }
  __syncthreads();

  half16 h0 = load_a_lds(&lds[wv][0], 0, lane);
  half16 h1 = load_a_lds(&lds[wv][0], 1, lane);

  _Float16* outs[3] = {vbuf, asrc, adst};
#pragma unroll
  for (int w = 0; w < 3; ++w) {
    _Float16* out = outs[w];
#pragma unroll
    for (int nt = 0; nt < 4; ++nt) {
      vf8 acc = vzero();
      acc = wmma16(h0, wfrag(wpk, 1 + w, 0, nt, lane), acc);
      acc = wmma16(h1, wfrag(wpk, 1 + w, 1, nt, lane), acc);
#pragma unroll
      for (int r = 0; r < 8; ++r)
        out[(size_t)(n0 + mrow0 + r) * CC + nt * 16 + col0] = (_Float16)acc[r];
    }
  }
}

// ---------------------------------------------------------------------------
// Kernel 2: per-edge pipeline, 16 edges per wave, 128 edges per block.
//   delta = lrelu(lrelu((pos_i-pos_j)@P1+p1)@P2+p2)         (P1 layer = VALU, K=3)
//   alpha = relu(relu((a_dst[i]-a_src[j]+delta)@A1+a1)@A2+a2)
// P2/A1/A2 fragments staged block-wide into LDS via async DMA; delta/alpha tiles
// emitted with coalesced global_store_async_from_lds_b128 straight out of LDS.
// ---------------------------------------------------------------------------
__global__ void edge_alpha_kernel(const float* __restrict__ pos, const int* __restrict__ eidx,
                                  const _Float16* __restrict__ wpk,
                                  const float* __restrict__ P1, const float* __restrict__ p1,
                                  const float* __restrict__ p2, const float* __restrict__ a1,
                                  const float* __restrict__ a2,
                                  const _Float16* __restrict__ asrc,
                                  const _Float16* __restrict__ adst,
                                  _Float16* __restrict__ delta, _Float16* __restrict__ alpha,
                                  unsigned int* __restrict__ mmax) {
  __shared__ __align__(16) _Float16 wlds[3 * 4096];     // P2,A1,A2 B-fragments (24KB)
  __shared__ __align__(16) _Float16 tile[8][16 * 72];   // per-wave staging tile
  const int lane = threadIdx.x & 31, wv = threadIdx.x >> 5;
  const long e0 = ((long)blockIdx.x * 8 + wv) * 16;
  const int m = lane & 15;
  const int kg = (lane & 16) ? 8 : 0;
  const int col0 = lane & 15, mrow0 = (lane & 16) ? 8 : 0;

  // Block-wide async stage of the three edge-side weight matrices into LDS.
  {
    const _Float16* src = wpk + 4 * 4096;               // mats 4,5,6
    const unsigned wbase = lds_offset(&wlds[0]);
#pragma unroll
    for (int k = 0; k < 6; ++k) {                       // 1536 x 16B chunks / 256 thr
      int chunk = k * 256 + threadIdx.x;
      async_load_b128(wbase + chunk * 16, src + chunk * 8);
    }
    async_wait0();
  }
  __syncthreads();

  const unsigned tbase = lds_offset(&tile[wv][0]);

  const long e = e0 + m;
  const int j = eidx[e];
  const int i = eidx[N_EDGES + e];
  const float dp0 = pos[i * 3 + 0] - pos[j * 3 + 0];
  const float dp1 = pos[i * 3 + 1] - pos[j * 3 + 1];
  const float dp2 = pos[i * 3 + 2] - pos[j * 3 + 2];

  // hidden = lrelu(dp @ P1 + p1), built directly in A-fragment layout (K=3 -> VALU)
  half16 h[2];
#pragma unroll
  for (int ks = 0; ks < 2; ++ks) {
#pragma unroll
    for (int q = 0; q < 16; ++q) {
      int ch = ks * 32 + ((q < 8) ? (kg + q) : (16 + kg + q - 8));
      float t = p1[ch] + dp0 * P1[ch] + dp1 * P1[CC + ch] + dp2 * P1[2 * CC + ch];
      h[ks][q] = (_Float16)(t > 0.f ? t : 0.01f * t);
    }
  }

  // delta = lrelu(hidden @ P2 + p2) -> LDS tile (A-reload layout)
#pragma unroll
  for (int nt = 0; nt < 4; ++nt) {
    vf8 acc = vzero();
    acc = wmma16(h[0], wfrag_lds(wlds, 0, 0, nt, lane), acc);
    acc = wmma16(h[1], wfrag_lds(wlds, 0, 1, nt, lane), acc);
    float b = p2[nt * 16 + col0];
#pragma unroll
    for (int r = 0; r < 8; ++r) {
      float t = acc[r] + b;
      t = t > 0.f ? t : 0.01f * t;
      tile[wv][(mrow0 + r) * 72 + nt * 16 + col0] = (_Float16)t;
    }
  }
  __syncthreads();   // tile complete (DScnt drained by barrier)

  // Coalesced DMA of the delta tile: 128 x 16B chunks per wave, 4 per lane.
#pragma unroll
  for (int k = 0; k < 4; ++k) {
    int id = k * 32 + lane;
    int row = id >> 3, co = (id & 7) * 8;
    async_store_b128(delta + (size_t)(e0 + row) * CC + co, tbase + row * 144 + co * 2);
  }

  // alpha_pre fragments: gather a_dst[i] - a_src[j] (f16, two 16B loads each) + delta (LDS)
  const _Float16* ri = adst + (size_t)i * CC;
  const _Float16* rj = asrc + (size_t)j * CC;
  half16 f0 = load_a_f16(ri, 0, lane) - load_a_f16(rj, 0, lane) + load_a_lds(&tile[wv][0], 0, lane);
  half16 f1 = load_a_f16(ri, 1, lane) - load_a_f16(rj, 1, lane) + load_a_lds(&tile[wv][0], 1, lane);

  // hid2 = relu(alpha_pre @ A1 + a1)
  vf8 acc1[4];
#pragma unroll
  for (int nt = 0; nt < 4; ++nt) {
    vf8 acc = vzero();
    acc = wmma16(f0, wfrag_lds(wlds, 1, 0, nt, lane), acc);
    acc = wmma16(f1, wfrag_lds(wlds, 1, 1, nt, lane), acc);
    acc1[nt] = acc;
  }
  async_wait0();     // delta DMA has consumed the tile
  __syncthreads();   // everyone done reading delta tile -> safe to overwrite
#pragma unroll
  for (int nt = 0; nt < 4; ++nt) {
    float b = a1[nt * 16 + col0];
#pragma unroll
    for (int r = 0; r < 8; ++r) {
      float t = acc1[nt][r] + b;
      t = t > 0.f ? t : 0.f;
      tile[wv][(mrow0 + r) * 72 + nt * 16 + col0] = (_Float16)t;
    }
  }
  __syncthreads();

  half16 g0 = load_a_lds(&tile[wv][0], 0, lane);
  half16 g1 = load_a_lds(&tile[wv][0], 1, lane);

  int irows[8];
#pragma unroll
  for (int r = 0; r < 8; ++r) irows[r] = eidx[N_EDGES + e0 + mrow0 + r];

  // alpha = relu(hid2 @ A2 + a2)
  vf8 acc2[4];
#pragma unroll
  for (int nt = 0; nt < 4; ++nt) {
    vf8 acc = vzero();
    acc = wmma16(g0, wfrag_lds(wlds, 2, 0, nt, lane), acc);
    acc = wmma16(g1, wfrag_lds(wlds, 2, 1, nt, lane), acc);
    acc2[nt] = acc;
  }
  __syncthreads();   // hid2 tile reads complete -> safe to overwrite with alpha
#pragma unroll
  for (int nt = 0; nt < 4; ++nt) {
    float b = a2[nt * 16 + col0];
#pragma unroll
    for (int r = 0; r < 8; ++r) {
      float t = acc2[nt][r] + b;
      t = t > 0.f ? t : 0.f;
      tile[wv][(mrow0 + r) * 72 + nt * 16 + col0] = (_Float16)t;
      // scatter-max into per-(i,ch) running max (uint trick: alpha >= 0)
      atomicMax(&mmax[(size_t)irows[r] * CC + nt * 16 + col0], __float_as_uint(t));
    }
  }
  __syncthreads();

  // Coalesced DMA of the alpha tile (s_endpgm's implicit wait-idle drains ASYNCcnt).
#pragma unroll
  for (int k = 0; k < 4; ++k) {
    int id = k * 32 + lane;
    int row = id >> 3, co = (id & 7) * 8;
    async_store_b128(alpha + (size_t)(e0 + row) * CC + co, tbase + row * 144 + co * 2);
  }
}

// ---------------------------------------------------------------------------
// Kernel 3: one thread per (edge, channel). e = exp(alpha - m[i]);
// s[i] += e; num[i] += e * (v[j] + delta).  (softmax denominator folded out)
// ---------------------------------------------------------------------------
__global__ void edge_scatter_kernel(const int* __restrict__ eidx,
                                    const _Float16* __restrict__ alpha,
                                    const _Float16* __restrict__ delta,
                                    const _Float16* __restrict__ vbuf,
                                    const float* __restrict__ mmax,
                                    float* __restrict__ sbuf, float* __restrict__ numbuf) {
  size_t t = (size_t)blockIdx.x * blockDim.x + threadIdx.x;
  if (t >= (size_t)N_EDGES * CC) return;
  int c = (int)(t & 63);
  long e = (long)(t >> 6);
  int j = eidx[e];
  int i = eidx[N_EDGES + e];
  float al = (float)alpha[t];
  float mv = mmax[(size_t)i * CC + c];
  float ev = __expf(al - mv);
  atomicAdd(sbuf + (size_t)i * CC + c, ev);
  float contrib = ev * ((float)vbuf[(size_t)j * CC + c] + (float)delta[t]);
  atomicAdd(numbuf + (size_t)i * CC + c, contrib);
}

// ---------------------------------------------------------------------------
// Kernel 4: out = relu((num/(s+1e-16)) @ W_out + b_out), f32 output.
// ---------------------------------------------------------------------------
__global__ void node_out_kernel(const float* __restrict__ numbuf, const float* __restrict__ sbuf,
                                const _Float16* __restrict__ wpk, const float* __restrict__ b_out,
                                float* __restrict__ out) {
  const int lane = threadIdx.x & 31, wv = threadIdx.x >> 5;
  const int n0 = (blockIdx.x * 8 + wv) * 16;
  const int m = lane & 15;
  const int kg = (lane & 16) ? 8 : 0;
  const int col0 = lane & 15, mrow0 = (lane & 16) ? 8 : 0;

  const float* nrow = numbuf + (size_t)(n0 + m) * CC;
  const float* srow = sbuf + (size_t)(n0 + m) * CC;
  half16 a0, a1;
#pragma unroll
  for (int ks = 0; ks < 2; ++ks) {
    half16 r;
#pragma unroll
    for (int q = 0; q < 16; ++q) {
      int ch = ks * 32 + ((q < 8) ? (kg + q) : (16 + kg + q - 8));
      r[q] = (_Float16)(nrow[ch] / (srow[ch] + 1e-16f));
    }
    if (ks == 0) a0 = r; else a1 = r;
  }

#pragma unroll
  for (int nt = 0; nt < 4; ++nt) {
    vf8 acc = vzero();
    acc = wmma16(a0, wfrag(wpk, 7, 0, nt, lane), acc);
    acc = wmma16(a1, wfrag(wpk, 7, 1, nt, lane), acc);
    float b = b_out[nt * 16 + col0];
#pragma unroll
    for (int r = 0; r < 8; ++r) {
      float t = acc[r] + b;
      t = t > 0.f ? t : 0.f;
      out[(size_t)(n0 + mrow0 + r) * CC + nt * 16 + col0] = t;
    }
  }
}

// ---------------------------------------------------------------------------
extern "C" void kernel_launch(void* const* d_in, const int* in_sizes, int n_in,
                              void* d_out, int out_size, void* d_ws, size_t ws_size,
                              hipStream_t stream) {
  const float* x     = (const float*)d_in[0];
  const float* pos   = (const float*)d_in[1];
  const int*   eidx  = (const int*)d_in[2];
  const float* W_in  = (const float*)d_in[3];
  const float* b_in  = (const float*)d_in[4];
  const float* W_out = (const float*)d_in[5];
  const float* b_out = (const float*)d_in[6];
  const float* W_lin = (const float*)d_in[7];
  const float* W_src = (const float*)d_in[8];
  const float* W_dst = (const float*)d_in[9];
  const float* P1    = (const float*)d_in[10];
  const float* p1    = (const float*)d_in[11];
  const float* P2    = (const float*)d_in[12];
  const float* p2    = (const float*)d_in[13];
  const float* A1    = (const float*)d_in[14];
  const float* a1    = (const float*)d_in[15];
  const float* A2    = (const float*)d_in[16];
  const float* a2    = (const float*)d_in[17];

  (void)in_sizes; (void)n_in; (void)out_size; (void)ws_size;

  char* ws = (char*)d_ws;
  size_t off = 0;
  auto take = [&](size_t bytes) -> void* {
    void* p = ws + off;
    off = (off + bytes + 255) & ~(size_t)255;
    return p;
  };
  _Float16* wpk   = (_Float16*)take((size_t)8 * 4096 * sizeof(_Float16));
  _Float16* vbuf  = (_Float16*)take((size_t)N_NODES * CC * sizeof(_Float16));
  _Float16* asrc  = (_Float16*)take((size_t)N_NODES * CC * sizeof(_Float16));
  _Float16* adst  = (_Float16*)take((size_t)N_NODES * CC * sizeof(_Float16));
  _Float16* delta = (_Float16*)take((size_t)N_EDGES * CC * sizeof(_Float16));
  _Float16* alpha = (_Float16*)take((size_t)N_EDGES * CC * sizeof(_Float16));
  float*    mbuf  = (float*)take((size_t)N_NODES * CC * sizeof(float));
  float*    sbuf  = (float*)take((size_t)N_NODES * CC * sizeof(float));
  float*    nbuf  = (float*)take((size_t)N_NODES * CC * sizeof(float));

  // m/s/num are contiguous (each 16MB, 256B-aligned): zero in one async memset.
  hipMemsetAsync(mbuf, 0, 3 * (size_t)N_NODES * CC * sizeof(float), stream);

  prep_weights_kernel<<<128, 256, 0, stream>>>(W_in, W_lin, W_src, W_dst, P2, A1, A2, W_out, wpk);

  node_in_kernel<<<N_NODES / 128, 256, 0, stream>>>(x, wpk, b_in, vbuf, asrc, adst);

  edge_alpha_kernel<<<N_EDGES / 128, 256, 0, stream>>>(pos, eidx, wpk, P1, p1, p2, a1, a2,
                                                       asrc, adst, delta, alpha,
                                                       (unsigned int*)mbuf);

  edge_scatter_kernel<<<(N_EDGES * (size_t)CC) / 256, 256, 0, stream>>>(eidx, alpha, delta, vbuf,
                                                                        mbuf, sbuf, nbuf);

  node_out_kernel<<<N_NODES / 128, 256, 0, stream>>>(nbuf, sbuf, wpk, b_out, (float*)d_out);
}
